// TokenPose_TB_base_61040075210880
// MI455X (gfx1250) — matrix operand order repaired and checked by reading, not compiled, not run on hardware
//
#include <hip/hip_runtime.h>
#include <hip/hip_bf16.h>
#include <cstdint>

// ---------------- constants ----------------
#define BATCH   16
#define SEQ     273
#define NPAD    288
#define DIM     768
#define HEADS   12
#define HD      64
#define MLPD    2304
#define DEPTHL  12
#define MROWS   (BATCH * SEQ)      // 4368
#define SCALE_Q 0.125f             // 64^-0.5

typedef __attribute__((ext_vector_type(16))) __bf16 v16bf;
typedef __attribute__((ext_vector_type(8)))  __bf16 v8bf;
typedef __attribute__((ext_vector_type(8)))  float  v8f;

// ---- fragment loaders (wave32 WMMA, contiguous k order used by A and B) ----
// A (16x32 bf16): lane L, row M=L%16; lanes<16 hold K 0..15 contiguous, lanes>=16 K 16..31.
// Caller passes p already offset by half*16.
__device__ inline v16bf load_a_frag(const __bf16* p) {
    union { v16bf v; v8bf h[2]; } u;
    u.h[0] = *(const v8bf*)(p);
    u.h[1] = *(const v8bf*)(p + 8);
    return u.v;
}
// B (32x16 bf16): lane L, col N=L%16; lanes<16 hold K 0..15 contiguous, lanes>=16 K 16..31.
// Caller passes p already offset by half*16.
__device__ inline v16bf load_b_frag(const __bf16* p) {
    union { v16bf v; v8bf h[2]; } u;
    u.h[0] = *(const v8bf*)(p);
    u.h[1] = *(const v8bf*)(p + 8);
    return u.v;
}

// ---- cross-lane helpers: VALU permlane when available, DS bpermute fallback ----
#if __has_builtin(__builtin_amdgcn_permlane16)
#define HAVE_PERMLANE 1
__device__ inline float perm_max_step(float v, unsigned lo, unsigned hi) {
    unsigned iv = __float_as_uint(v);
    unsigned p  = __builtin_amdgcn_permlane16(iv, iv, lo, hi, false, false);
    return fmaxf(v, __uint_as_float(p));
}
// max across each 16-lane half (xor network 1,2,4,8)
__device__ inline float rowmax16(float v) {
    v = perm_max_step(v, 0x67452301u, 0xEFCDAB89u); // xor 1
    v = perm_max_step(v, 0x54761032u, 0xDCFE98BAu); // xor 2
    v = perm_max_step(v, 0x32107654u, 0xBA98FEDCu); // xor 4
    v = perm_max_step(v, 0xFEDCBA98u, 0x76543210u); // xor 8
    return v;
}
// broadcast lane 0 of each 16-lane half
__device__ inline float bcast16(float v) {
    unsigned iv = __float_as_uint(v);
    unsigned p  = __builtin_amdgcn_permlane16(iv, iv, 0u, 0u, false, false);
    return __uint_as_float(p);
}
#else
#define HAVE_PERMLANE 0
__device__ inline float rowmax16(float v) {
    for (int off = 1; off < 16; off <<= 1) v = fmaxf(v, __shfl_xor(v, off, 32));
    return v;
}
__device__ inline float bcast16(float v) {
    return __shfl(v, (threadIdx.x & 16), 32);
}
#endif

// ---- async global->LDS copy (gfx1250 DMA path) with sync fallback ----
// Probe result (round 2 diagnostic): param 0 is AS1 pointer to 'int __vector_size__(16)'.
#if __has_builtin(__builtin_amdgcn_global_load_async_to_lds_b128)
#define HAVE_ASYNC_LDS 1
typedef int v4i_vs __attribute__((vector_size(16)));
typedef __attribute__((address_space(1))) v4i_vs g_v4i;
typedef __attribute__((address_space(3))) v4i_vs l_v4i;
__device__ inline void async_cp16(const __bf16* g, __bf16* l) {
    __builtin_amdgcn_global_load_async_to_lds_b128(
        (g_v4i*)(uintptr_t)g,
        (l_v4i*)(unsigned)(uintptr_t)l,
        0, 0);
}
__device__ inline void wait_async0() {
#if __has_builtin(__builtin_amdgcn_s_wait_asynccnt)
    __builtin_amdgcn_s_wait_asynccnt(0);
#else
    asm volatile("s_wait_asynccnt 0" ::: "memory");
#endif
}
#else
#define HAVE_ASYNC_LDS 0
#endif

// ---------------- transpose + fp32->bf16 convert:  out[C][R] = (bf16) in[R][C] ----------------
__global__ void tconv_kernel(const float* __restrict__ in, __bf16* __restrict__ out, int R, int C) {
    __shared__ float tile[32][33];
    int c0 = blockIdx.x * 32, r0 = blockIdx.y * 32;
    int tx = threadIdx.x, ty = threadIdx.y;   // 32 x 8
#pragma unroll
    for (int i = 0; i < 32; i += 8) {
        int r = r0 + ty + i, c = c0 + tx;
        if (r < R && c < C) tile[ty + i][tx] = in[(size_t)r * C + c];
    }
    __syncthreads();
#pragma unroll
    for (int i = 0; i < 32; i += 8) {
        int c = c0 + ty + i, r = r0 + tx;
        if (r < R && c < C) out[(size_t)c * R + r] = (__bf16)tile[tx][ty + i];
    }
}

// ---------------- positional add: x[:, 17:, :] += pos * prod(la[1..idx]) ----------------
__global__ void posadd_kernel(float* __restrict__ x, const float* __restrict__ pos,
                              const float* __restrict__ la, int idx) {
    int t = blockIdx.x * blockDim.x + threadIdx.x;   // covers 16*256*768
    int c = t % DIM;
    int p = (t / DIM) % 256;
    int b = t / (DIM * 256);
    float sc = 1.0f;
    for (int i = 1; i <= idx; ++i) sc *= la[i];
    x[((size_t)(b * SEQ + 17 + p)) * DIM + c] += pos[(size_t)p * DIM + c] * sc;
}

// ---------------- LayerNorm (768 cols) -> bf16 ----------------
__global__ void ln_kernel(const float* __restrict__ x, const float* __restrict__ s,
                          const float* __restrict__ b, __bf16* __restrict__ out) {
    int row = blockIdx.x, t = threadIdx.x;     // 256 threads
    const float* xr = x + (size_t)row * DIM;
    float v0 = xr[t], v1 = xr[t + 256], v2 = xr[t + 512];
    __shared__ float s1[256], s2[256];
    s1[t] = v0 + v1 + v2;
    s2[t] = v0 * v0 + v1 * v1 + v2 * v2;
    __syncthreads();
    for (int off = 128; off > 0; off >>= 1) {
        if (t < off) { s1[t] += s1[t + off]; s2[t] += s2[t + off]; }
        __syncthreads();
    }
    float mean = s1[0] * (1.0f / DIM);
    float var  = s2[0] * (1.0f / DIM) - mean * mean;
    float inv  = rsqrtf(var + 1e-5f);
    __bf16* orow = out + (size_t)row * DIM;
    orow[t]       = (__bf16)((v0 - mean) * inv * s[t]       + b[t]);
    orow[t + 256] = (__bf16)((v1 - mean) * inv * s[t + 256] + b[t + 256]);
    orow[t + 512] = (__bf16)((v2 - mean) * inv * s[t + 512] + b[t + 512]);
}

// ---------------- bf16 WMMA GEMM with double-buffered LDS staging ----------------
// C[M,Nc] = A[M,K] * Bt[Nc,K]^T.  Block: 128x128 tile, 8 waves, K chunks of 64.
// EPI 0: plain fp32 out.  EPI 1: +bias +residual, fp32 out.  EPI 2: +bias, exact GELU, bf16 out.
template <int EPI>
__global__ __launch_bounds__(256) void gemm_bf16_kernel(
        const __bf16* __restrict__ A, const __bf16* __restrict__ Bt,
        float* __restrict__ Cf, __bf16* __restrict__ Cb,
        const float* __restrict__ bias, const float* __restrict__ res,
        int M, int K, int Nc) {
    constexpr int STR = 72;                       // padded LDS row stride (elements)
    __shared__ __bf16 As[2][128 * STR];
    __shared__ __bf16 Bs[2][128 * STR];
    int t = threadIdx.x;
    int wave = t >> 5, lane = t & 31, l16 = lane & 15, half = lane >> 4;
    int mblk = blockIdx.x * 128, nblk = blockIdx.y * 128;

    // cooperative staging: thread t owns 32 elements (64B) of row (t>>1), half (t&1)
    int li = t >> 1;
    int lo = (t & 1) * 32;
    int arow = mblk + li; if (arow >= M) arow = M - 1;     // clamp; stores guarded later
    const __bf16* agp = A  + (size_t)arow * K + lo;
    const __bf16* bgp = Bt + (size_t)(nblk + li) * K + lo;
    const int lbase = li * STR + lo;
    const int nk = K >> 6;                                  // K / 64

    v8f acc[8] = {};

    auto compute = [&](int buf) {
#pragma unroll
        for (int kk = 0; kk < 64; kk += 32) {
            v16bf a = load_a_frag(&As[buf][(wave * 16 + l16) * STR + kk + half * 16]);
#pragma unroll
            for (int nt = 0; nt < 8; ++nt) {
                v16bf b = load_b_frag(&Bs[buf][(nt * 16 + l16) * STR + kk + half * 16]);
                acc[nt] = __builtin_amdgcn_wmma_f32_16x16x32_bf16(
                              false, a, false, b, (short)0, acc[nt], false, false);
            }
        }
    };

#if HAVE_ASYNC_LDS
    auto issue = [&](int buf, int kc) {
        const __bf16* ga = agp + kc * 64;
        const __bf16* gb = bgp + kc * 64;
        __bf16* la = &As[buf][lbase];
        __bf16* lb = &Bs[buf][lbase];
#pragma unroll
        for (int q = 0; q < 4; ++q) {
            async_cp16(ga + q * 8, la + q * 8);
            async_cp16(gb + q * 8, lb + q * 8);
        }
    };
    issue(0, 0);
    wait_async0();
    __syncthreads();
    for (int kc = 0; kc < nk; ++kc) {
        int cur = kc & 1;
        if (kc + 1 < nk) issue(cur ^ 1, kc + 1);   // DMA overlaps with WMMA below
        compute(cur);
        if (kc + 1 < nk) { wait_async0(); __syncthreads(); }
    }
#else
    v8bf ra[4], rb[4];
    auto loadregs = [&](int kc) {
        const __bf16* ga = agp + kc * 64;
        const __bf16* gb = bgp + kc * 64;
#pragma unroll
        for (int q = 0; q < 4; ++q) {
            ra[q] = *(const v8bf*)(ga + q * 8);
            rb[q] = *(const v8bf*)(gb + q * 8);
        }
    };
    auto storesmem = [&](int buf) {
#pragma unroll
        for (int q = 0; q < 4; ++q) {
            *(v8bf*)(&As[buf][lbase + q * 8]) = ra[q];
            *(v8bf*)(&Bs[buf][lbase + q * 8]) = rb[q];
        }
    };
    loadregs(0); storesmem(0);
    __syncthreads();
    for (int kc = 0; kc < nk; ++kc) {
        int cur = kc & 1;
        if (kc + 1 < nk) loadregs(kc + 1);         // global loads in flight during compute
        compute(cur);
        if (kc + 1 < nk) { storesmem(cur ^ 1); __syncthreads(); }
    }
#endif

    // epilogue
#pragma unroll
    for (int nt = 0; nt < 8; ++nt) {
        int gcol = nblk + nt * 16 + l16;
        float bv = (EPI > 0) ? bias[gcol] : 0.0f;
#pragma unroll
        for (int r = 0; r < 8; ++r) {
            int grow = mblk + wave * 16 + r + half * 8;
            if (grow < M) {
                float val = acc[nt][r] + bv;
                size_t o = (size_t)grow * Nc + gcol;
                if (EPI == 2) {
                    val = 0.5f * val * (1.0f + erff(val * 0.70710678f));
                    Cb[o] = (__bf16)val;
                } else if (EPI == 1) {
                    Cf[o] = res[o] + val;
                } else {
                    Cf[o] = val;
                }
            }
        }
    }
}

// ---------------- RoPE + head split:  qkv fp32 -> Qb (scaled), Kb, Vt  (bf16, N padded) ----------------
__global__ void rope_kernel(const float* __restrict__ qkv,
                            const float* __restrict__ fc, const float* __restrict__ fs,
                            __bf16* __restrict__ Qb, __bf16* __restrict__ Kb,
                            __bf16* __restrict__ Vt, int layer) {
    int d  = threadIdx.x;                 // 0..63
    int n  = blockIdx.x % NPAD;
    int bh = blockIdx.x / NPAD;           // 0..191
    int b = bh / HEADS, h = bh % HEADS;
    size_t qidx = ((size_t)bh * NPAD + n) * HD + d;
    size_t vidx = ((size_t)bh * HD + d) * NPAD + n;
    if (n >= SEQ) { Qb[qidx] = (__bf16)0.0f; Kb[qidx] = (__bf16)0.0f; Vt[vidx] = (__bf16)0.0f; return; }
    const float* base = qkv + ((size_t)(b * SEQ + n)) * (3 * DIM) + h * HD;
    int   p  = d & ~1;
    size_t fi = (((size_t)layer * HEADS + h) * SEQ + n) * (HD / 2) + (d >> 1);
    float c = fc[fi], sn = fs[fi];
    float qr = base[p], qi = base[p + 1];
    float q = (d & 1) ? (qr * sn + qi * c) : (qr * c - qi * sn);
    float kr = base[DIM + p], ki = base[DIM + p + 1];
    float k = (d & 1) ? (kr * sn + ki * c) : (kr * c - ki * sn);
    float v = base[2 * DIM + d];
    Qb[qidx] = (__bf16)(q * SCALE_Q);
    Kb[qidx] = (__bf16)k;
    Vt[vidx] = (__bf16)v;
}

// ---------------- flash attention, WMMA bf16, one wave per 16-row Q tile ----------------
// Row-sums of P come from a 5th WMMA tile whose B-fragment is a constant ones-column.
__global__ __launch_bounds__(128) void attn_kernel(const __bf16* __restrict__ Qb,
                                                   const __bf16* __restrict__ Kb,
                                                   const __bf16* __restrict__ Vt,
                                                   __bf16* __restrict__ Obf) {
    __shared__ __bf16 Plds[4][16 * 32];
    int wave = threadIdx.x >> 5, lane = threadIdx.x & 31;
    int l16 = lane & 15, half = lane >> 4;
    int bh = blockIdx.x / 5;
    int qt = (blockIdx.x % 5) * 4 + wave;
    if (qt >= NPAD / 16) return;                       // whole-wave exit (EXEC stays full for WMMA)
    int b = bh / HEADS, h = bh % HEADS;

    const __bf16* qrow = Qb + ((size_t)bh * NPAD + qt * 16 + l16) * HD;
    v16bf qa0 = load_a_frag(qrow + half * 16);          // d = 0..31
    v16bf qa1 = load_a_frag(qrow + 32 + half * 16);     // d = 32..63

    // constant B-fragment: column 0 of the 5th tile is all-ones -> P row-sums
    __bf16 ov = (l16 == 0) ? (__bf16)1.0f : (__bf16)0.0f;
    v16bf ones;
#pragma unroll
    for (int i = 0; i < 16; ++i) ones[i] = ov;

    v8f oacc[5] = {};                                  // 4 d-tiles + row-sum tile
    float rm[8];
#pragma unroll
    for (int r = 0; r < 8; ++r) rm[r] = -1e30f;
    __bf16* pl = Plds[wave];

    for (int jb = 0; jb < NPAD; jb += 32) {
        const __bf16* krow0 = Kb + ((size_t)bh * NPAD + jb + l16) * HD;
        const __bf16* krow1 = krow0 + 16 * HD;
        v8f s0 = {}, s1 = {};
        s0 = __builtin_amdgcn_wmma_f32_16x16x32_bf16(false, qa0, false, load_b_frag(krow0 + half * 16),      (short)0, s0, false, false);
        s0 = __builtin_amdgcn_wmma_f32_16x16x32_bf16(false, qa1, false, load_b_frag(krow0 + 32 + half * 16), (short)0, s0, false, false);
        s1 = __builtin_amdgcn_wmma_f32_16x16x32_bf16(false, qa0, false, load_b_frag(krow1 + half * 16),      (short)0, s1, false, false);
        s1 = __builtin_amdgcn_wmma_f32_16x16x32_bf16(false, qa1, false, load_b_frag(krow1 + 32 + half * 16), (short)0, s1, false, false);

        // mask padded key columns (this lane owns one column per S tile)
        if (jb + l16 >= SEQ) {
#pragma unroll
            for (int r = 0; r < 8; ++r) s0[r] = -1e30f;
        }
        if (jb + 16 + l16 >= SEQ) {
#pragma unroll
            for (int r = 0; r < 8; ++r) s1[r] = -1e30f;
        }

        float nm[8], corr[8];
#pragma unroll
        for (int r = 0; r < 8; ++r) {
            float tmax = rowmax16(fmaxf(s0[r], s1[r]));
            nm[r]   = fmaxf(rm[r], tmax);
            corr[r] = __expf(rm[r] - nm[r]);
            rm[r]   = nm[r];
        }
#pragma unroll
        for (int dt = 0; dt < 5; ++dt)
#pragma unroll
            for (int r = 0; r < 8; ++r) oacc[dt][r] *= corr[r];
#pragma unroll
        for (int r = 0; r < 8; ++r) {
            float p0 = __expf(s0[r] - nm[r]);
            float p1 = __expf(s1[r] - nm[r]);
            pl[(r + half * 8) * 32 + l16]      = (__bf16)p0;   // DS in-order within wave
            pl[(r + half * 8) * 32 + 16 + l16] = (__bf16)p1;
        }
        v16bf pa = load_a_frag(pl + l16 * 32 + half * 16);      // P as 16x32 A-fragment
#pragma unroll
        for (int dt = 0; dt < 4; ++dt) {
            const __bf16* vrow = Vt + ((size_t)bh * HD + dt * 16 + l16) * NPAD + jb + half * 16;
            oacc[dt] = __builtin_amdgcn_wmma_f32_16x16x32_bf16(
                           false, pa, false, load_b_frag(vrow), (short)0, oacc[dt], false, false);
        }
        oacc[4] = __builtin_amdgcn_wmma_f32_16x16x32_bf16(
                      false, pa, false, ones, (short)0, oacc[4], false, false);
    }

    // recover row-sums l from column 0 of tile 4 (lives in lane 0 / lane 16)
    float linv[8];
#pragma unroll
    for (int r = 0; r < 8; ++r) linv[r] = 1.0f / bcast16(oacc[4][r]);

#pragma unroll
    for (int dt = 0; dt < 4; ++dt)
#pragma unroll
        for (int r = 0; r < 8; ++r) {
            int m = qt * 16 + r + half * 8;
            if (m < SEQ)
                Obf[((size_t)b * SEQ + m) * DIM + h * HD + dt * 16 + l16] =
                    (__bf16)(oacc[dt][r] * linv[r]);
        }
}

// ---------------- host orchestration ----------------
extern "C" void kernel_launch(void* const* d_in, const int* in_sizes, int n_in,
                              void* d_out, int out_size, void* d_ws, size_t ws_size,
                              hipStream_t stream) {
    const float* in_x    = (const float*)d_in[0];
    const float* fc      = (const float*)d_in[1];
    const float* fs      = (const float*)d_in[2];
    const float* la      = (const float*)d_in[3];
    const float* pos     = (const float*)d_in[4];
    const float* ln1_s   = (const float*)d_in[5];
    const float* ln1_b   = (const float*)d_in[6];
    const float* w_qkv   = (const float*)d_in[7];
    const float* w_out   = (const float*)d_in[8];
    const float* b_out   = (const float*)d_in[9];
    const float* ln2_s   = (const float*)d_in[10];
    const float* ln2_b   = (const float*)d_in[11];
    const float* w1      = (const float*)d_in[12];
    const float* b1      = (const float*)d_in[13];
    const float* w2      = (const float*)d_in[14];
    const float* b2      = (const float*)d_in[15];

    float* x = (float*)d_out;                     // running activations live in d_out

    // workspace layout (bytes)
    char* ws = (char*)d_ws;
    size_t off = 0;
    auto alloc = [&](size_t bytes) { char* p = ws + off; off += (bytes + 255) & ~(size_t)255; return p; };
    __bf16* WqkvT = (__bf16*)alloc((size_t)3 * DIM * DIM * 2);           // [2304, 768]
    __bf16* WoutT = (__bf16*)alloc((size_t)DIM * DIM * 2);               // [768, 768]
    __bf16* W1T   = (__bf16*)alloc((size_t)MLPD * DIM * 2);              // [2304, 768]
    __bf16* W2T   = (__bf16*)alloc((size_t)DIM * MLPD * 2);              // [768, 2304]
    __bf16* h_bf  = (__bf16*)alloc((size_t)MROWS * DIM * 2);
    float*  qkv   = (float*) alloc((size_t)MROWS * 3 * DIM * 4);
    __bf16* Qb    = (__bf16*)alloc((size_t)BATCH * HEADS * NPAD * HD * 2);
    __bf16* Kb    = (__bf16*)alloc((size_t)BATCH * HEADS * NPAD * HD * 2);
    __bf16* Vt    = (__bf16*)alloc((size_t)BATCH * HEADS * HD * NPAD * 2);
    __bf16* o_bf  = (__bf16*)alloc((size_t)MROWS * DIM * 2);
    __bf16* ff_bf = (__bf16*)alloc((size_t)MROWS * MLPD * 2);
    (void)ws_size; (void)in_sizes; (void)n_in; (void)out_size;

    (void)hipMemcpyAsync(x, in_x, (size_t)MROWS * DIM * sizeof(float),
                         hipMemcpyDeviceToDevice, stream);

    dim3 tcb(32, 8);
    const int GX = (MROWS + 127) / 128;          // 35

    for (int idx = 0; idx < DEPTHL; ++idx) {
        if (idx > 0)
            posadd_kernel<<<(BATCH * 256 * DIM) / 256, 256, 0, stream>>>(x, pos, la, idx);

        // per-layer weight transpose+convert to bf16 (W^T row-major -> contiguous WMMA B fragments)
        tconv_kernel<<<dim3(3 * DIM / 32, DIM / 32), tcb, 0, stream>>>(
            w_qkv + (size_t)idx * DIM * 3 * DIM, WqkvT, DIM, 3 * DIM);
        tconv_kernel<<<dim3(DIM / 32, DIM / 32), tcb, 0, stream>>>(
            w_out + (size_t)idx * DIM * DIM, WoutT, DIM, DIM);
        tconv_kernel<<<dim3(MLPD / 32, DIM / 32), tcb, 0, stream>>>(
            w1 + (size_t)idx * DIM * MLPD, W1T, DIM, MLPD);
        tconv_kernel<<<dim3(DIM / 32, MLPD / 32), tcb, 0, stream>>>(
            w2 + (size_t)idx * MLPD * DIM, W2T, MLPD, DIM);

        // LN1 -> bf16
        ln_kernel<<<MROWS, 256, 0, stream>>>(x, ln1_s + idx * DIM, ln1_b + idx * DIM, h_bf);

        // qkv = h @ Wqkv   [4368 x 2304]
        gemm_bf16_kernel<0><<<dim3(GX, 3 * DIM / 128), 256, 0, stream>>>(
            h_bf, WqkvT, qkv, nullptr, nullptr, nullptr, MROWS, DIM, 3 * DIM);

        // RoPE + head split + V transpose, zero-padded to NPAD
        rope_kernel<<<BATCH * HEADS * NPAD, HD, 0, stream>>>(qkv, fc, fs, Qb, Kb, Vt, idx);

        // flash attention -> o (bf16)
        attn_kernel<<<BATCH * HEADS * 5, 128, 0, stream>>>(Qb, Kb, Vt, o_bf);

        // x = x + o @ Wout + b_out
        gemm_bf16_kernel<1><<<dim3(GX, DIM / 128), 256, 0, stream>>>(
            o_bf, WoutT, x, nullptr, b_out + idx * DIM, x, MROWS, DIM, DIM);

        // LN2 -> bf16
        ln_kernel<<<MROWS, 256, 0, stream>>>(x, ln2_s + idx * DIM, ln2_b + idx * DIM, h_bf);

        // ff = gelu(h2 @ W1 + b1)   (bf16 out)
        gemm_bf16_kernel<2><<<dim3(GX, MLPD / 128), 256, 0, stream>>>(
            h_bf, W1T, nullptr, ff_bf, b1 + idx * MLPD, nullptr, MROWS, DIM, MLPD);

        // x = x + ff @ W2 + b2
        gemm_bf16_kernel<1><<<dim3(GX, DIM / 128), 256, 0, stream>>>(
            ff_bf, W2T, x, nullptr, b2 + idx * DIM, x, MROWS, MLPD, DIM);
    }
}
